// SparseMHA_30545807409987
// MI455X (gfx1250) — matrix operand order repaired and checked by reading, not compiled, not run on hardware
//
#include <hip/hip_runtime.h>
#include <hip/hip_bf16.h>

#define NN     4096
#define INDIM  512
#define DD     64
#define HH     8

typedef __bf16 bf16_t;
typedef __attribute__((ext_vector_type(16))) __bf16 v16bf;
typedef __attribute__((ext_vector_type(8)))  __bf16 v8bf;
typedef __attribute__((ext_vector_type(8)))  float  v8f;
typedef __attribute__((ext_vector_type(4)))  unsigned int v4u;
typedef __attribute__((ext_vector_type(8)))  int v8i;
typedef __attribute__((ext_vector_type(4)))  int v4i;

union V16U { v16bf v; struct { v8bf lo, hi; } p; };

// ---------------------------------------------------------------------------
// TDM: async-load one 32x32 f32 tile of the adjacency matrix into LDS.
// D# built per ISA ch.8: group0 = {count=1, lds_addr, global_addr, type=2},
// group1 = {data_size=4B, tensor_dim0/1 = 4096, tile_dim0/1 = 32,
//           tensor_dim0_stride = 4096}. Tracked with TENSORcnt.
// ---------------------------------------------------------------------------
__device__ __forceinline__ void tdm_load_tile_a(const float* gsrc, unsigned lds_off)
{
    const unsigned long long ga = (unsigned long long)(uintptr_t)gsrc;
    v4u g0;
    g0[0] = 1u;                                        // count=1 (valid user D#)
    g0[1] = lds_off;                                   // lds_addr (bytes)
    g0[2] = (unsigned)(ga & 0xFFFFFFFFu);              // global_addr[31:0]
    g0[3] = (unsigned)((ga >> 32) & 0x01FFFFFFu)       // global_addr[56:32]
          | 0x80000000u;                               // type=2 (bits 127:126)
    v8i g1;
    g1[0] = (int)(2u << 16);                           // data_size = 4 bytes
    g1[1] = (int)(((unsigned)NN & 0xFFFFu) << 16);     // tensor_dim0[15:0]
    g1[2] = (int)((((unsigned)NN >> 16) & 0xFFFFu)     // tensor_dim0[31:16]
          | (((unsigned)NN & 0xFFFFu) << 16));         // tensor_dim1[15:0]
    g1[3] = (int)((((unsigned)NN >> 16) & 0xFFFFu)     // tensor_dim1[31:16]
          | (32u << 16));                              // tile_dim0 = 32
    g1[4] = 32;                                        // tile_dim1 = 32 (dim2=0)
    g1[5] = NN;                                        // tensor_dim0_stride[31:0]
    g1[6] = 0;
    g1[7] = 0;
    v4i z = {0, 0, 0, 0};
#if defined(__clang_major__) && (__clang_major__ >= 23)
    v8i z8 = {0, 0, 0, 0, 0, 0, 0, 0};
    __builtin_amdgcn_tensor_load_to_lds(g0, g1, z, z, z8, 0);
#else
    __builtin_amdgcn_tensor_load_to_lds(g0, g1, z, z, 0);
#endif
}

// ---------------------------------------------------------------------------
// Kernel 1: f32 -> bf16 conversion. h stays row-major; W is stored TRANSPOSED
// (Wt[mat][c][k]) so GEMM B-operands become one contiguous 32B load per lane.
// ---------------------------------------------------------------------------
__global__ void cvt_kernel(const float* __restrict__ h,
                           const float* __restrict__ Wq,
                           const float* __restrict__ Wk,
                           const float* __restrict__ Wv,
                           bf16_t* __restrict__ hb,
                           bf16_t* __restrict__ Wt)
{
    const int HTOT = NN * INDIM;            // 2,097,152
    const int WTOT = 3 * INDIM * INDIM;     //   786,432
    for (int i = blockIdx.x * blockDim.x + threadIdx.x;
         i < HTOT + WTOT; i += gridDim.x * blockDim.x) {
        if (i < HTOT) {
            hb[i] = (bf16_t)h[i];
        } else {
            int j   = i - HTOT;
            int mat = j / (INDIM * INDIM);
            int rem = j - mat * INDIM * INDIM;
            int k   = rem / INDIM;          // input-dim row of W
            int c   = rem - k * INDIM;      // output column
            const float* W = (mat == 0) ? Wq : (mat == 1) ? Wk : Wv;
            Wt[mat * INDIM * INDIM + c * INDIM + k] = (bf16_t)W[k * INDIM + c];
        }
    }
}

// ---------------------------------------------------------------------------
// Kernel 2: QKV projection GEMM. grid=(N/16, 3). 8 waves/block, each wave owns
// 64 output columns (4x 16-wide WMMA tiles), K-loop of 32 over INDIM=512,
// unrolled 4x so >=16 global_load_b128 can be in flight across WMMA issues.
// Epilogue: +bias, Q*=1/8, scatter into head-split bf16 layouts:
//   Qh/Kh[h][n][d] row-major,  Vt[h][d][n] (transposed for AV B-operand).
// ---------------------------------------------------------------------------
__global__ __launch_bounds__(256) void qkv_kernel(
    const bf16_t* __restrict__ hb, const bf16_t* __restrict__ Wt,
    const float* __restrict__ bq, const float* __restrict__ bk,
    const float* __restrict__ bv,
    bf16_t* __restrict__ Qh, bf16_t* __restrict__ Kh, bf16_t* __restrict__ Vt)
{
    const int mat  = blockIdx.y;
    const int rn   = blockIdx.x * 16;
    const int lane = threadIdx.x & 31;
    const int wave = threadIdx.x >> 5;
    const int lh   = lane >> 4;     // lane half (ISA A/B operand swizzle)
    const int l16  = lane & 15;

    const bf16_t* arow = hb + (size_t)(rn + l16) * INDIM;
    const bf16_t* wb   = Wt + (size_t)mat * INDIM * INDIM;
    const float*  bias = (mat == 0) ? bq : (mat == 1) ? bk : bv;

    v8f vzero = {};
    v8f acc[4];
#pragma unroll
    for (int ct = 0; ct < 4; ++ct) acc[ct] = vzero;

#pragma unroll 4
    for (int kk = 0; kk < INDIM; kk += 32) {
        // A operand 16x32 bf16: lane<16 holds K {kk..kk+15},
        // lane>=16 holds K {kk+16..kk+31}  (two 16B loads)
        const int ka = kk + (lh ? 16 : 0);
        V16U a;
        a.p.lo = *(const v8bf*)(arow + ka);
        a.p.hi = *(const v8bf*)(arow + ka + 8);
        // B operand 32x16: lane holds column (l16), contiguous 16 K values
        const int kb = kk + (lh ? 16 : 0);
        v16bf b[4];
#pragma unroll
        for (int ct = 0; ct < 4; ++ct) {
            const int c = wave * 64 + ct * 16 + l16;
            b[ct] = *(const v16bf*)(wb + (size_t)c * INDIM + kb);
        }
#pragma unroll
        for (int ct = 0; ct < 4; ++ct)
            acc[ct] = __builtin_amdgcn_wmma_f32_16x16x32_bf16(
                false, a.v, false, b[ct], (short)0, acc[ct], false, false);
    }

#pragma unroll
    for (int ct = 0; ct < 4; ++ct) {
        const int c    = wave * 64 + ct * 16 + l16;
        const int hidx = c & 7;          // reshape(N, 64, 8): head is fastest
        const int d    = c >> 3;
        const float bc = bias[c];
#pragma unroll
        for (int r = 0; r < 8; ++r) {
            const int n = rn + r + (lh << 3);   // C layout: lane>=16 -> M+8
            float val = acc[ct][r] + bc;
            if (mat == 0)
                Qh[(size_t)hidx * NN * DD + (size_t)n * DD + d] = (bf16_t)(val * 0.125f);
            else if (mat == 1)
                Kh[(size_t)hidx * NN * DD + (size_t)n * DD + d] = (bf16_t)val;
            else
                Vt[(size_t)hidx * DD * NN + (size_t)d * NN + n] = (bf16_t)val;
        }
    }
}

// ---------------------------------------------------------------------------
// Kernel 3: attention with softmax over the HEAD axis.
// grid = N/32 blocks of 256 threads; wave w owns head w. Per m-tile of 32:
//   wave0 TDM-prefetches the NEXT A tile (double-buffered LDS, TENSORcnt),
//   S = Q*K^T (8 WMMA) -> LDS -> cooperative mask*softmax over 8 heads
//   (A==0 rows correctly become uniform 1/8) -> bf16 weights in LDS ->
//   O += W*V (8 WMMA, V pre-transposed). O stays in registers (64 VGPR/lane).
// ---------------------------------------------------------------------------
__global__ __launch_bounds__(256) void attn_kernel(
    const float* __restrict__ Adj,
    const bf16_t* __restrict__ Qh, const bf16_t* __restrict__ Kh,
    const bf16_t* __restrict__ Vt, float* __restrict__ out)
{
    __shared__ __align__(16) float  Slds[HH][32][32];   // 32 KB
    __shared__ __align__(16) bf16_t Wlds[HH][32][32];   // 16 KB
    __shared__ __align__(16) float  Alds[2][32][32];    //  8 KB (TDM dst)

    const int tid  = threadIdx.x;
    const int lane = tid & 31;
    const int h    = tid >> 5;          // wave == head
    const int lh   = lane >> 4;
    const int l16  = lane & 15;
    const int n0   = blockIdx.x * 32;

    // Kick off TDM for the first adjacency tile (wave 0 only: one DMA per WG)
    if (h == 0)
        tdm_load_tile_a(Adj + (size_t)n0 * NN,
                        (unsigned)(uintptr_t)&Alds[0][0][0]);

    // Persistent Q A-operands: 2 row-subtiles x 2 K-chunks (K=64 total)
    v16bf aQ[2][2];
#pragma unroll
    for (int nt = 0; nt < 2; ++nt) {
        const bf16_t* qrow = Qh + (size_t)h * NN * DD + (size_t)(n0 + nt * 16 + l16) * DD;
#pragma unroll
        for (int kc = 0; kc < 2; ++kc) {
            const int ka = kc * 32 + (lh ? 16 : 0);
            V16U a;
            a.p.lo = *(const v8bf*)(qrow + ka);
            a.p.hi = *(const v8bf*)(qrow + ka + 8);
            aQ[nt][kc] = a.v;
        }
    }

    v8f vzero = {};
    v8f o[2][4];
#pragma unroll
    for (int nt = 0; nt < 2; ++nt)
#pragma unroll
        for (int dt = 0; dt < 4; ++dt) o[nt][dt] = vzero;

    for (int m0 = 0; m0 < NN; m0 += 32) {
        const int buf = (m0 >> 5) & 1;
        // Prefetch NEXT A tile into the other buffer while we compute
        if (h == 0 && m0 + 32 < NN)
            tdm_load_tile_a(Adj + (size_t)n0 * NN + (m0 + 32),
                            (unsigned)(uintptr_t)&Alds[buf ^ 1][0][0]);

        // ---- S = Q * K^T for this wave's head (loads hoisted) -------------
        v16bf bK[2][2];
#pragma unroll
        for (int mt = 0; mt < 2; ++mt) {
            const bf16_t* krow =
                Kh + (size_t)h * NN * DD + (size_t)(m0 + mt * 16 + l16) * DD;
#pragma unroll
            for (int kc = 0; kc < 2; ++kc)
                bK[mt][kc] = *(const v16bf*)(krow + kc * 32 + (lh ? 16 : 0));
        }
        v8f s[2][2];
#pragma unroll
        for (int nt = 0; nt < 2; ++nt)
#pragma unroll
            for (int mt = 0; mt < 2; ++mt) s[nt][mt] = vzero;
#pragma unroll
        for (int kc = 0; kc < 2; ++kc)
#pragma unroll
            for (int mt = 0; mt < 2; ++mt)
#pragma unroll
                for (int nt = 0; nt < 2; ++nt)
                    s[nt][mt] = __builtin_amdgcn_wmma_f32_16x16x32_bf16(
                        false, aQ[nt][kc], false, bK[mt][kc], (short)0,
                        s[nt][mt], false, false);

        // ---- scatter S tile to LDS (C layout: VGPR r -> row, lane -> col) -
#pragma unroll
        for (int nt = 0; nt < 2; ++nt)
#pragma unroll
            for (int mt = 0; mt < 2; ++mt)
#pragma unroll
                for (int r = 0; r < 8; ++r)
                    Slds[h][nt * 16 + r + (lh << 3)][mt * 16 + l16] = s[nt][mt][r];

        // current A tile must have landed; in-order TDM => cnt<=1 suffices
        if (h == 0) {
            if (m0 + 32 < NN) __builtin_amdgcn_s_wait_tensorcnt(1);
            else              __builtin_amdgcn_s_wait_tensorcnt(0);
        }
        __syncthreads();

        // ---- mask + softmax over the 8 heads (1024 elems / 256 threads) ---
#pragma unroll
        for (int e = tid; e < 1024; e += 256) {
            const int i = e >> 5, j = e & 31;
            const float a = Alds[buf][i][j];
            float sc[HH];
            float mx = -3.0e38f;
#pragma unroll
            for (int hh = 0; hh < HH; ++hh) {
                sc[hh] = Slds[hh][i][j] * a;   // A==0 -> all zero -> w = 1/8
                mx = fmaxf(mx, sc[hh]);
            }
            float sum = 0.f;
#pragma unroll
            for (int hh = 0; hh < HH; ++hh) {
                sc[hh] = __expf(sc[hh] - mx);
                sum += sc[hh];
            }
            const float inv = __builtin_amdgcn_rcpf(sum);
#pragma unroll
            for (int hh = 0; hh < HH; ++hh)
                Wlds[hh][i][j] = (bf16_t)(sc[hh] * inv);
        }
        __syncthreads();

        // ---- O += W * V   (A from LDS, B from transposed V in global) -----
        v16bf aW[2];
#pragma unroll
        for (int nt = 0; nt < 2; ++nt) {
            const int row  = nt * 16 + l16;
            const int base = lh ? 16 : 0;
            V16U a;
            a.p.lo = *(const v8bf*)&Wlds[h][row][base];
            a.p.hi = *(const v8bf*)&Wlds[h][row][base + 8];
            aW[nt] = a.v;
        }
        v16bf bV[4];
#pragma unroll
        for (int dt = 0; dt < 4; ++dt) {
            const bf16_t* vrow = Vt + (size_t)h * DD * NN +
                                 (size_t)(dt * 16 + l16) * NN + m0 + (lh ? 16 : 0);
            bV[dt] = *(const v16bf*)vrow;
        }
#pragma unroll
        for (int dt = 0; dt < 4; ++dt)
#pragma unroll
            for (int nt = 0; nt < 2; ++nt)
                o[nt][dt] = __builtin_amdgcn_wmma_f32_16x16x32_bf16(
                    false, aW[nt], false, bV[dt], (short)0, o[nt][dt], false, false);
        __syncthreads();   // protect Wlds/Slds before next iteration
    }

    // ---- epilogue: out is (H, N, D) flattened (== reference reshape) ------
#pragma unroll
    for (int nt = 0; nt < 2; ++nt)
#pragma unroll
        for (int dt = 0; dt < 4; ++dt)
#pragma unroll
            for (int r = 0; r < 8; ++r) {
                const int n = n0 + nt * 16 + r + (lh << 3);
                const int d = dt * 16 + l16;
                out[(size_t)h * NN * DD + (size_t)n * DD + d] = o[nt][dt][r];
            }
}

// ---------------------------------------------------------------------------
extern "C" void kernel_launch(void* const* d_in, const int* in_sizes, int n_in,
                              void* d_out, int out_size, void* d_ws, size_t ws_size,
                              hipStream_t stream)
{
    const float* A  = (const float*)d_in[0];
    const float* h  = (const float*)d_in[1];
    const float* Wq = (const float*)d_in[2];
    const float* bq = (const float*)d_in[3];
    const float* Wk = (const float*)d_in[4];
    const float* bk = (const float*)d_in[5];
    const float* Wv = (const float*)d_in[6];
    const float* bv = (const float*)d_in[7];
    float* out = (float*)d_out;

    // workspace layout (~18.3 MB total)
    char* ws = (char*)d_ws;
    bf16_t* hb = (bf16_t*)ws;                                  // 4 MB
    bf16_t* Wt = (bf16_t*)(ws + (size_t)4 * 1024 * 1024);      // 1.5 MB
    bf16_t* Qh = (bf16_t*)(ws + (size_t)4 * 1024 * 1024 + 1536 * 1024);
    bf16_t* Kh = Qh + (size_t)HH * NN * DD;                    // +4 MB
    bf16_t* Vt = Kh + (size_t)HH * NN * DD;                    // +4 MB

    cvt_kernel<<<2048, 256, 0, stream>>>(h, Wq, Wk, Wv, hb, Wt);
    qkv_kernel<<<dim3(NN / 16, 3), 256, 0, stream>>>(hb, Wt, bq, bk, bv, Qh, Kh, Vt);
    attn_kernel<<<NN / 32, 256, 0, stream>>>(A, Qh, Kh, Vt, out);
}
